// EncoderRNN_16123307229637
// MI455X (gfx1250) — compile-verified, matrix operand order checked by reading, not run on hardware
//
#include <hip/hip_runtime.h>
#include <hip/hip_bf16.h>
#include <stdint.h>

#define T_LEN 1024
#define B_N   64
#define H_N   256
#define G_N   768   // 3*H
#define TB    (T_LEN*B_N)

typedef __attribute__((ext_vector_type(16))) __bf16 v16bf;
typedef __attribute__((ext_vector_type(8)))  float  v8f;

union ABf   { uint4 u[2]; v16bf v; };
union U16x8 { unsigned short s[8]; uint4 u; };

__device__ __forceinline__ unsigned short f2bf(float f) {
  union { float f; unsigned u; } x; x.f = f;
  unsigned r = x.u + 0x7FFFu + ((x.u >> 16) & 1u);   // RNE
  return (unsigned short)(r >> 16);
}
__device__ __forceinline__ float bf2f(unsigned short b) {
  union { unsigned u; float f; } x; x.u = ((unsigned)b) << 16;
  return x.f;
}
__device__ __forceinline__ float sigmoidf_(float x) {
  return 1.0f / (1.0f + __expf(-x));
}

// ---------------- utility kernels ----------------

__global__ void zero_f32(float* __restrict__ p, size_t n) {
  size_t i = (size_t)blockIdx.x * blockDim.x + threadIdx.x;
  size_t stride = (size_t)gridDim.x * blockDim.x;
  for (; i < n; i += stride) p[i] = 0.0f;
}

__global__ void cvt_bf16(const float* __restrict__ src,
                         unsigned short* __restrict__ dst, int n) {
  int i = blockIdx.x * blockDim.x + threadIdx.x;
  if (i < n) dst[i] = f2bf(src[i]);
}

__global__ void embed_gather(const int* __restrict__ seq,
                             const float* __restrict__ emb,
                             unsigned short* __restrict__ embBF) {
  int i = blockIdx.x;                 // 0..TB-1
  int tok = seq[i];
  float v = emb[(size_t)tok * H_N + threadIdx.x];
  embBF[(size_t)i * H_N + threadIdx.x] = f2bf(v);
}

// ---------------- xg = emb @ Wih^T + bih (both dirs) ----------------
// Output stored in WMMA C-fragment layout:
//   xgT[d][mTile_g (4096)][nTile (48)][lane (32)][r (8)]  (bf16)
// so the scan kernel can reload accumulator-shaped tiles with b128 loads.

__global__ void __launch_bounds__(256)
xg_gemm(const unsigned short* __restrict__ embBF,
        const unsigned short* __restrict__ WihF,
        const unsigned short* __restrict__ WihB,
        const float* __restrict__ bihF,
        const float* __restrict__ bihB,
        unsigned short* __restrict__ xgT) {
  const int lane = threadIdx.x & 31;
  const int wave = threadIdx.x >> 5;
  const int JOBS_PER_DIR = 4096 * 12;
  int job = blockIdx.x * 8 + wave;
  int d   = job / JOBS_PER_DIR;
  int rem = job - d * JOBS_PER_DIR;
  int mT  = rem / 12;
  int nG  = rem - mT * 12;

  const unsigned short* W = d ? WihB : WihF;
  const float* bih        = d ? bihB : bihF;

  const int nlane = lane & 15;
  const int klo   = (lane < 16) ? 0 : 8;   // ISA 16-bit A/B K-pattern
  const int arow  = mT * 16 + nlane;

  // preload all A fragments (one emb row per lane, 512B)
  ABf a[8];
  #pragma unroll
  for (int kk = 0; kk < 8; ++kk) {
    const unsigned short* ap = embBF + (size_t)arow * H_N + kk*32 + klo;
    a[kk].u[0] = *(const uint4*)(ap);
    a[kk].u[1] = *(const uint4*)(ap + 16);
  }

  const v8f vz = {0.f,0.f,0.f,0.f,0.f,0.f,0.f,0.f};
  v8f acc[4];
  #pragma unroll
  for (int t = 0; t < 4; ++t) acc[t] = vz;

  #pragma unroll
  for (int kk = 0; kk < 8; ++kk) {
    const int koff = kk * 32 + klo;
    #pragma unroll
    for (int t = 0; t < 4; ++t) {
      ABf b;
      const unsigned short* bp = W + (size_t)(nG*64 + t*16 + nlane) * H_N + koff;
      b.u[0] = *(const uint4*)(bp);
      b.u[1] = *(const uint4*)(bp + 16);
      acc[t] = __builtin_amdgcn_wmma_f32_16x16x32_bf16(false, a[kk].v, false, b.v,
                                                       (short)0, acc[t], false, false);
    }
  }

  // store in C-fragment layout: 4 coalesced b128 stores per lane
  unsigned short* dst = xgT + ((((size_t)d*4096 + mT)*48 + nG*4) << 8) + lane*8;
  #pragma unroll
  for (int t = 0; t < 4; ++t) {
    const float bj = bih[nG*64 + t*16 + nlane];
    U16x8 o;
    #pragma unroll
    for (int r = 0; r < 8; ++r) o.s[r] = f2bf(acc[t][r] + bj);
    *(uint4*)(dst + (size_t)t*256) = o.u;
  }
}

// ---------------- recurrent scan ----------------
// grid = 8 WGs: dir = blk>>2 (0=fwd,1=bwd), slice s = blk&3 (64 h-columns).
// Whh slice pinned in LDS; h staged into LDS each step via
// global_load_async_to_lds_b128; step barrier = per-direction atomic counter.

#define WHH_PITCH 264             // bf16 elems per LDS row (bank-conflict pad)
#define LDS_WHH_BYTES (192*WHH_PITCH*2)
#define LDS_HS_BYTES  (64*68*4)
#define LDS_HA_BYTES  (64*WHH_PITCH*2)

__global__ void __launch_bounds__(128)
gru_scan(const unsigned short* __restrict__ xgT,    // C-frag layout, bf16
         const unsigned short* __restrict__ WhhF,
         const unsigned short* __restrict__ WhhB,
         const float* __restrict__ bhhF,
         const float* __restrict__ bhhB,
         const int* __restrict__ lengths,
         unsigned short* __restrict__ hbuf,         // [2][2][64][256] bf16
         unsigned int* __restrict__ ctr,            // [2]
         float* __restrict__ out)                   // [T][B][H] ++ [2][B][H]
{
  extern __shared__ char smem[];
  unsigned short* whh_s = (unsigned short*)smem;                       // [192][264]
  float*          h_s   = (float*)(smem + LDS_WHH_BYTES);              // [64][68]
  unsigned short* hA_s  = (unsigned short*)(smem + LDS_WHH_BYTES + LDS_HS_BYTES); // [64][264]

  const int tid  = threadIdx.x;
  const int lane = tid & 31;
  const int mT   = tid >> 5;            // wave id == M tile (16 batch rows)
  const int d = blockIdx.x >> 2;
  const int s = blockIdx.x & 3;

  const unsigned short* Whh = d ? WhhB : WhhF;
  const float* bhh          = d ? bhhB : bhhF;

  // preload Whh slice: gate g rows [g*H + 64s, g*H + 64s + 64)
  for (int idx = tid; idx < 192*32; idx += 128) {
    int lr = idx >> 5;
    int c8 = (idx & 31) * 8;
    int grow = (lr >> 6) * H_N + s*64 + (lr & 63);
    *(uint4*)(whh_s + lr*WHH_PITCH + c8) = *(const uint4*)(Whh + (size_t)grow * H_N + c8);
  }
  for (int idx = tid; idx < 64*68; idx += 128) h_s[idx] = 0.0f;

  const int nlane = lane & 15;
  const int klo = (lane < 16) ? 0 : 8;
  const int hi8 = (lane < 16) ? 0 : 8;
  int lenr[8];
  #pragma unroll
  for (int r = 0; r < 8; ++r) lenr[r] = lengths[mT*16 + r + hi8];
  float br_[4], bz_[4], bn_[4];
  #pragma unroll
  for (int t = 0; t < 4; ++t) {
    int c = t*16 + nlane;
    br_[t] = bhh[          s*64 + c];
    bz_[t] = bhh[H_N     + s*64 + c];
    bn_[t] = bhh[2*H_N   + s*64 + c];
  }
  __syncthreads();

  const int arow = mT*16 + nlane;
  // ISA 10.2: low 32 bits of a flat LDS address are the LDS byte offset.
  const unsigned haLds = (unsigned)(uintptr_t)hA_s;
  unsigned int* myctr = ctr + d;

  #pragma unroll 1
  for (int step = 0; step < T_LEN; ++step) {
    const int p = step & 1;
    const int tstep = d ? (T_LEN - 1 - step) : step;
    const unsigned short* hA = hbuf + (size_t)(d*2 + p)     * (64*H_N);
    unsigned short*       hW = hbuf + (size_t)(d*2 + (p^1)) * (64*H_N);

    // prefetch this step's xg tile-blocks (3 gates x 2KB, 48 x 128B lines)
    const char* pf = (const char*)(xgT + ((((size_t)d*4096 + (size_t)tstep*4 + mT)*48) << 8));
    #pragma unroll
    for (int q = 0; q < 2; ++q) {
      int l = lane + q*32;
      if (l < 48) {
        int g = l >> 4, lr = l & 15;
        __builtin_prefetch(pf + ((size_t)(16*g + 4*s)*256 + lr*64)*2, 0, 1);
      }
    }

    // async-stage this wave's 16 h rows into LDS (bf16, padded pitch)
    #pragma unroll
    for (int i = 0; i < 16; ++i) {
      const int row = mT*16 + i;
      unsigned lofs = haLds + (unsigned)((row*WHH_PITCH + lane*8) * 2);
      const unsigned short* gp = hA + (size_t)row * H_N + lane*8;
      asm volatile("global_load_async_to_lds_b128 %0, %1, off"
                   :: "v"(lofs), "v"((unsigned long long)(uintptr_t)gp)
                   : "memory");
    }
    asm volatile("s_wait_asynccnt 0x0" ::: "memory");

    const v8f vz = {0.f,0.f,0.f,0.f,0.f,0.f,0.f,0.f};
    v8f acc[12];
    #pragma unroll
    for (int t = 0; t < 12; ++t) acc[t] = vz;

    // hg = h @ Whh_slice^T   (64 x 256 x 192; A and B both from LDS)
    #pragma unroll
    for (int kk = 0; kk < 8; ++kk) {
      const int koff = kk*32 + klo;
      ABf a;
      const unsigned short* ap = hA_s + arow*WHH_PITCH + koff;
      a.u[0] = *(const uint4*)(ap);
      a.u[1] = *(const uint4*)(ap + 16);
      #pragma unroll
      for (int t = 0; t < 12; ++t) {
        ABf b;
        const unsigned short* bp = whh_s + (t*16 + nlane)*WHH_PITCH + koff;
        b.u[0] = *(const uint4*)(bp);
        b.u[1] = *(const uint4*)(bp + 16);
        acc[t] = __builtin_amdgcn_wmma_f32_16x16x32_bf16(false, a.v, false, b.v,
                                                         (short)0, acc[t], false, false);
      }
    }

    // gates: xg tiles come back in accumulator layout (b128 per gate-tile)
    const unsigned short* xtb = xgT
        + ((((size_t)d*4096 + (size_t)tstep*4 + mT)*48) << 8) + (size_t)lane*8;
    #pragma unroll
    for (int t = 0; t < 4; ++t) {
      U16x8 xr8, xz8, xn8;
      xr8.u = *(const uint4*)(xtb + (size_t)(      4*s + t)*256);
      xz8.u = *(const uint4*)(xtb + (size_t)(16 +  4*s + t)*256);
      xn8.u = *(const uint4*)(xtb + (size_t)(32 +  4*s + t)*256);
      const int c = t*16 + nlane;
      #pragma unroll
      for (int r = 0; r < 8; ++r) {
        const int brow = mT*16 + r + hi8;
        float rg = sigmoidf_(bf2f(xr8.s[r]) + acc[t][r]   + br_[t]);
        float zg = sigmoidf_(bf2f(xz8.s[r]) + acc[4+t][r] + bz_[t]);
        float ng = tanhf(bf2f(xn8.s[r]) + rg * (acc[8+t][r] + bn_[t]));
        float hp = h_s[brow*68 + c];
        float hn = (1.0f - zg) * ng + zg * hp;
        bool m = tstep < lenr[r];
        float hx = m ? hn : hp;
        h_s[brow*68 + c] = hx;
        hW[(size_t)brow * H_N + s*64 + c] = f2bf(hx);
        if (m) atomicAdd(&out[((size_t)tstep * B_N + brow) * H_N + s*64 + c], hn);
      }
    }

    // per-direction step barrier across 4 WGs
    __threadfence();
    __syncthreads();
    if (tid == 0) {
      __hip_atomic_fetch_add(myctr, 1u, __ATOMIC_RELEASE, __HIP_MEMORY_SCOPE_AGENT);
      unsigned int tgt = 4u * (unsigned)(step + 1);
      while (__hip_atomic_load(myctr, __ATOMIC_ACQUIRE, __HIP_MEMORY_SCOPE_AGENT) < tgt) {
        __builtin_amdgcn_s_sleep(1);
      }
    }
    __syncthreads();
  }

  // final hidden: out[T*B*H + d*B*H + b*H + (64s + c)]
  for (int idx = tid; idx < 64*64; idx += 128) {
    int brow = idx >> 6;
    int c = idx & 63;
    out[(size_t)T_LEN * B_N * H_N + (size_t)d * B_N * H_N
        + (size_t)brow * H_N + s*64 + c] = h_s[brow*68 + c];
  }
}

// ---------------- launch ----------------

extern "C" void kernel_launch(void* const* d_in, const int* in_sizes, int n_in,
                              void* d_out, int out_size, void* d_ws, size_t ws_size,
                              hipStream_t stream) {
  const int*   seq  = (const int*)d_in[0];
  const int*   lens = (const int*)d_in[1];
  const float* emb  = (const float*)d_in[2];
  const float* WihF = (const float*)d_in[3];
  const float* WhhF = (const float*)d_in[4];
  const float* bihF = (const float*)d_in[5];
  const float* bhhF = (const float*)d_in[6];
  const float* WihB = (const float*)d_in[7];
  const float* WhhB = (const float*)d_in[8];
  const float* bihB = (const float*)d_in[9];
  const float* bhhB = (const float*)d_in[10];
  float* out = (float*)d_out;

  char* ws = (char*)d_ws;
  unsigned short* embBF = (unsigned short*)(ws);                    // 33,554,432 B
  unsigned short* WihFb = (unsigned short*)(ws + 33554432);         // 4 x 393,216 B
  unsigned short* WhhFb = WihFb + 196608;
  unsigned short* WihBb = WhhFb + 196608;
  unsigned short* WhhBb = WihBb + 196608;
  unsigned short* xgT   = (unsigned short*)(ws + 35127296);         // 201,326,592 B
  unsigned short* hbuf  = (unsigned short*)(ws + 236453888);        // 262,144 B
  unsigned int*   ctr   = (unsigned int*)(ws + 236716032);          // counters

  // zero atomically-accumulated output, h ping-pong buffers, counters
  zero_f32<<<2048, 256, 0, stream>>>(out, (size_t)out_size);
  zero_f32<<<64, 256, 0, stream>>>((float*)hbuf, (size_t)((262144 + 256) / 4));

  cvt_bf16<<<768, 256, 0, stream>>>(WihF, WihFb, 196608);
  cvt_bf16<<<768, 256, 0, stream>>>(WhhF, WhhFb, 196608);
  cvt_bf16<<<768, 256, 0, stream>>>(WihB, WihBb, 196608);
  cvt_bf16<<<768, 256, 0, stream>>>(WhhB, WhhBb, 196608);

  embed_gather<<<TB, 256, 0, stream>>>(seq, emb, embBF);

  xg_gemm<<<12288, 256, 0, stream>>>(embBF, WihFb, WihBb, bihF, bihB, xgT);

  size_t shm = LDS_WHH_BYTES + LDS_HS_BYTES + LDS_HA_BYTES;  // 152,576 B
  gru_scan<<<8, 128, shm, stream>>>(xgT, WhhFb, WhhBb, bhhF, bhhB, lens,
                                    hbuf, ctr, out);
}
